// AspectSent_27754078667134
// MI455X (gfx1250) — matrix operand order repaired and checked by reading, not compiled, not run on hardware
//
#include <hip/hip_runtime.h>

// ---------------- problem constants ----------------
#define BB 128
#define TT 100
#define BT (BB*TT)
#define ED 300
#define MD 50
#define D0P 352        // 350 padded to multiple of 32
#define HH 256
#define HX 512
#define G3 768         // 3*HH

typedef __attribute__((ext_vector_type(16))) _Float16 v16h;
typedef __attribute__((ext_vector_type(8)))  _Float16 v8h;
typedef __attribute__((ext_vector_type(8)))  float    v8f;

// Fragment loaders (CDNA5 ISA 7.12.2 layouts).
// A (16-bit, 16x32): lane half 0 -> K {k0..k0+7, k0+16..k0+23}, half 1 -> +8.
__device__ __forceinline__ v16h load_afrag(const _Float16* p) {
  v8h lo = *(const v8h*)p;
  v8h hi = *(const v8h*)(p + 16);
  v16h r;
#pragma unroll
  for (int e = 0; e < 8; ++e) { r[e] = lo[e]; r[8 + e] = hi[e]; }
  return r;
}
// B (16-bit, 32x16): lane = column, same K order as A: half 0 -> {k0..k0+7, k0+16..k0+23}, half 1 -> +8.
__device__ __forceinline__ v16h load_bfrag(const _Float16* p) {
  v8h b0 = *(const v8h*)p;
  v8h b1 = *(const v8h*)(p + 16);
  v16h r;
#pragma unroll
  for (int e = 0; e < 8; ++e) { r[e] = b0[e]; r[8 + e] = b1[e]; }
  return r;
}

// =====================================================================
// Generic WMMA GEMM, 2x2 register-tiled (32x32 output block per wave):
//   C[M,N] (+)= A[rowmap[m], K] * W[N,K]^T
// rowmap==nullptr -> identity; rowmap[m]==-1 -> row reads from ZROW (zeros),
// keeping all loads unconditional (EXEC stays full, no per-iter exec masks).
// =====================================================================
__global__ __launch_bounds__(256) void wmma_gemm_kernel(
    const _Float16* __restrict__ A, int lda,
    const _Float16* __restrict__ W, int ldw,
    float* __restrict__ Cf, _Float16* __restrict__ Ch, int ldc,
    const int* __restrict__ rowmap, const _Float16* __restrict__ zrow,
    int Ntiles2, int numTiles2, int Kchunks, int accumulate)
{
  int lane = threadIdx.x & 31;
  int wave = threadIdx.x >> 5;
  int tile = blockIdx.x * 8 + wave;
  if (tile >= numTiles2) return;                // wave-uniform
  int mt2 = tile / Ntiles2, nt2 = tile % Ntiles2;
  int m0 = mt2 * 32, n0 = nt2 * 32;
  int hl  = lane >> 4;                          // lane half (0/1)
  int l16 = lane & 15;

  int am0 = m0 + l16, am1 = m0 + 16 + l16;
  int ar0 = rowmap ? rowmap[am0] : am0;
  int ar1 = rowmap ? rowmap[am1] : am1;
  const _Float16* ab0 = (ar0 >= 0) ? (A + (size_t)ar0 * lda) : zrow;
  const _Float16* ab1 = (ar1 >= 0) ? (A + (size_t)ar1 * lda) : zrow;
  const _Float16* wb0 = W + (size_t)(n0 + l16) * ldw;
  const _Float16* wb1 = W + (size_t)(n0 + 16 + l16) * ldw;

  v8f a00 = {}, a01 = {}, a10 = {}, a11 = {};
  if (accumulate) {
#pragma unroll
    for (int r = 0; r < 8; ++r) {
      size_t row = (size_t)(m0 + r + hl * 8) * ldc;
      size_t rw1 = (size_t)(m0 + 16 + r + hl * 8) * ldc;
      a00[r] = Cf[row + n0 + l16];
      a01[r] = Cf[row + n0 + 16 + l16];
      a10[r] = Cf[rw1 + n0 + l16];
      a11[r] = Cf[rw1 + n0 + 16 + l16];
    }
  }

  for (int kc = 0; kc < Kchunks; ++kc) {
    int ka = kc * 32 + hl * 8;                  // A-side base
    int kb = kc * 32 + hl * 8;                  // B-side base
    __builtin_prefetch(wb0 + kb + 32, 0, 1);
    __builtin_prefetch(wb1 + kb + 32, 0, 1);
    v16h fa0 = load_afrag(ab0 + ka);
    v16h fa1 = load_afrag(ab1 + ka);
    v16h fb0 = load_bfrag(wb0 + kb);
    v16h fb1 = load_bfrag(wb1 + kb);
    a00 = __builtin_amdgcn_wmma_f32_16x16x32_f16(false, fa0, false, fb0, (short)0, a00, false, false);
    a01 = __builtin_amdgcn_wmma_f32_16x16x32_f16(false, fa0, false, fb1, (short)0, a01, false, false);
    a10 = __builtin_amdgcn_wmma_f32_16x16x32_f16(false, fa1, false, fb0, (short)0, a10, false, false);
    a11 = __builtin_amdgcn_wmma_f32_16x16x32_f16(false, fa1, false, fb1, (short)0, a11, false, false);
  }

#pragma unroll
  for (int r = 0; r < 8; ++r) {
    size_t row = (size_t)(m0 + r + hl * 8) * ldc;
    size_t rw1 = (size_t)(m0 + 16 + r + hl * 8) * ldc;
    if (Ch) {
      Ch[row + n0 + l16]      = (_Float16)a00[r];
      Ch[row + n0 + 16 + l16] = (_Float16)a01[r];
      Ch[rw1 + n0 + l16]      = (_Float16)a10[r];
      Ch[rw1 + n0 + 16 + l16] = (_Float16)a11[r];
    } else {
      Cf[row + n0 + l16]      = a00[r];
      Cf[row + n0 + 16 + l16] = a01[r];
      Cf[rw1 + n0 + l16]      = a10[r];
      Cf[rw1 + n0 + 16 + l16] = a11[r];
    }
  }
}

// =====================================================================
// Persistent bidirectional GRU layer.  grid=2 (dir), block=1024 (32 waves),
// dynamic LDS = 128*256 f16 = 64KB hidden state.
// Per step: GH = h @ whh^T via 2x2-tiled WMMA (96 macro-tiles, 3/wave),
// then fused gates.  Backward dir writes output already un-reversed.
// =====================================================================
__global__ __launch_bounds__(1024) void gru_layer_kernel(
    const _Float16* __restrict__ GI,    // [2][BT][768] input projections (f16)
    const _Float16* __restrict__ WHH,   // [2][768][256] f16 (this layer)
    const float* __restrict__ bih_f, const float* __restrict__ bhh_f,
    const float* __restrict__ bih_b, const float* __restrict__ bhh_b,
    const int* __restrict__ lens,
    _Float16* __restrict__ XO,          // [BT][512]  (fwd cols 0..255, bwd 256..511)
    float* __restrict__ GH,             // [2][128][768] scratch
    float* __restrict__ HF)             // [2][128][256] f32 hidden
{
  extern __shared__ _Float16 hsh[];     // 128*256 f16 hidden state
  const int dir = blockIdx.x;
  const _Float16* gi  = GI  + (size_t)dir * BT * G3;
  const _Float16* whh = WHH + (size_t)dir * G3 * HH;
  const float* bih = dir ? bih_b : bih_f;
  const float* bhh = dir ? bhh_b : bhh_f;
  float* gh = GH + (size_t)dir * BB * G3;
  float* hf = HF + (size_t)dir * BB * HH;

  const int tid = threadIdx.x;
  for (int i = tid; i < BB * HH; i += 1024) { hsh[i] = (_Float16)0.f; hf[i] = 0.f; }
  __syncthreads();

  const int lane = tid & 31, wave = tid >> 5;
  const int hl = lane >> 4, l16 = lane & 15;

  for (int t = 0; t < TT; ++t) {
    // ---- phase 1: GH = h @ whh^T  (M=128, N=768, K=256), 32x32 blocks ----
    for (int i = 0; i < 3; ++i) {
      int tile = wave * 3 + i;                  // 0..95
      int mt2 = tile / 24, nt2 = tile % 24;
      int m0 = mt2 * 32, n0 = nt2 * 32;
      const _Float16* wb0 = whh + (size_t)(n0 + l16) * HH;
      const _Float16* wb1 = whh + (size_t)(n0 + 16 + l16) * HH;
      v8f a00 = {}, a01 = {}, a10 = {}, a11 = {};
#pragma unroll
      for (int kc = 0; kc < 8; ++kc) {
        int ka = kc * 32 + hl * 8;
        int kb = kc * 32 + hl * 8;
        v16h fa0, fa1, fb0, fb1;
        {
          int b0i = (m0 + l16) * HH + ka;       // LDS, 16B aligned
          int b1i = (m0 + 16 + l16) * HH + ka;
          v8h lo0 = *(const v8h*)&hsh[b0i];
          v8h hi0 = *(const v8h*)&hsh[b0i + 16];
          v8h lo1 = *(const v8h*)&hsh[b1i];
          v8h hi1 = *(const v8h*)&hsh[b1i + 16];
#pragma unroll
          for (int e = 0; e < 8; ++e) {
            fa0[e] = lo0[e]; fa0[8 + e] = hi0[e];
            fa1[e] = lo1[e]; fa1[8 + e] = hi1[e];
          }
        }
        fb0 = load_bfrag(wb0 + kb);
        fb1 = load_bfrag(wb1 + kb);
        a00 = __builtin_amdgcn_wmma_f32_16x16x32_f16(false, fa0, false, fb0, (short)0, a00, false, false);
        a01 = __builtin_amdgcn_wmma_f32_16x16x32_f16(false, fa0, false, fb1, (short)0, a01, false, false);
        a10 = __builtin_amdgcn_wmma_f32_16x16x32_f16(false, fa1, false, fb0, (short)0, a10, false, false);
        a11 = __builtin_amdgcn_wmma_f32_16x16x32_f16(false, fa1, false, fb1, (short)0, a11, false, false);
      }
#pragma unroll
      for (int r = 0; r < 8; ++r) {
        int row = (m0 + r + hl * 8) * G3;
        int rw1 = (m0 + 16 + r + hl * 8) * G3;
        gh[row + n0 + l16]      = a00[r];
        gh[row + n0 + 16 + l16] = a01[r];
        gh[rw1 + n0 + l16]      = a10[r];
        gh[rw1 + n0 + 16 + l16] = a11[r];
      }
    }
    __threadfence();
    __syncthreads();

    // ---- phase 2: fused gates + state update + output scatter ----
    for (int e = tid; e < BB * HH; e += 1024) {
      int b_ = e >> 8, j = e & 255;
      size_t gr = ((size_t)b_ * TT + t) * G3;
      float gir = (float)gi[gr + j]           + bih[j];
      float giz = (float)gi[gr + 256 + j]     + bih[256 + j];
      float gin = (float)gi[gr + 512 + j]     + bih[512 + j];
      float ghr = gh[b_ * G3 + j]             + bhh[j];
      float ghz = gh[b_ * G3 + 256 + j]       + bhh[256 + j];
      float ghn = gh[b_ * G3 + 512 + j]       + bhh[512 + j];
      float r = 1.f / (1.f + __expf(-(gir + ghr)));
      float z = 1.f / (1.f + __expf(-(giz + ghz)));
      float n = tanhf(gin + r * ghn);
      float hold = hf[e];
      float hnew = (1.f - z) * n + z * hold;
      int L = lens[b_];
      bool valid = (t < L);
      float hsel = valid ? hnew : hold;
      hf[e] = hsel;
      hsh[e] = (_Float16)hsel;
      int pos = dir ? (valid ? (L - 1 - t) : t) : t;   // un-reverse backward dir
      XO[((size_t)b_ * TT + pos) * HX + dir * HH + j] =
          (_Float16)(valid ? hnew : 0.f);
    }
    __threadfence();
    __syncthreads();
  }
}

// ---------------- small helper kernels ----------------
__global__ void zero16_kernel(_Float16* __restrict__ p, long n) {
  long i = (long)blockIdx.x * blockDim.x + threadIdx.x;
  if (i < n) p[i] = (_Float16)0.f;
}

__global__ void pack16_kernel(const float* __restrict__ src, _Float16* __restrict__ dst,
                              int rows, int ksrc, int kdst) {
  long i = (long)blockIdx.x * blockDim.x + threadIdx.x;
  long total = (long)rows * kdst;
  if (i >= total) return;
  int r = (int)(i / kdst), k = (int)(i % kdst);
  dst[i] = (k < ksrc) ? (_Float16)src[(size_t)r * ksrc + k] : (_Float16)0.f;
}

__global__ void packconv_kernel(const float* __restrict__ cw, _Float16* __restrict__ dst) {
  long i = (long)blockIdx.x * blockDim.x + threadIdx.x;
  if (i >= 3L * HX * HX) return;
  int dt = (int)(i / (HX * HX));
  int rem = (int)(i % (HX * HX));
  int o = rem / HX, ci = rem % HX;
  dst[i] = (_Float16)cw[((size_t)o * HX + ci) * 3 + dt];
}

__global__ void embed_kernel(const int* __restrict__ sents, const int* __restrict__ masks,
                             const float* __restrict__ wemb, const float* __restrict__ memb,
                             _Float16* __restrict__ X) {
  long i = (long)blockIdx.x * blockDim.x + threadIdx.x;
  if (i >= (long)BT * D0P) return;
  int m = (int)(i / D0P), k = (int)(i % D0P);
  float v = 0.f;
  if (k < ED)            v = wemb[(size_t)sents[m] * ED + k];
  else if (k < ED + MD)  v = memb[masks[m] * MD + (k - ED)];
  X[i] = (_Float16)v;
}

__global__ void build_maps_kernel(const int* __restrict__ lens,
                                  int* __restrict__ rev, int* __restrict__ cm0,
                                  int* __restrict__ cm2) {
  int m = blockIdx.x * blockDim.x + threadIdx.x;
  if (m >= BT) return;
  int b = m / TT, t = m % TT;
  int L = lens[b];
  rev[m] = b * TT + (t < L ? (L - 1 - t) : t);
  cm0[m] = (t == 0)      ? -1 : m - 1;
  cm2[m] = (t == TT - 1) ? -1 : m + 1;
}

__global__ void conv_epilogue_kernel(float* __restrict__ ctx, const float* __restrict__ cb) {
  long i = (long)blockIdx.x * blockDim.x + threadIdx.x;
  if (i >= (long)BT * HX) return;
  float v = ctx[i] + cb[i % HX];
  ctx[i] = v > 0.f ? v : 0.f;
}

__global__ void avg_kernel(const float* __restrict__ ctx, const int* __restrict__ masks,
                           float* __restrict__ avg) {
  int b = blockIdx.x, h = threadIdx.x;
  float s = 0.f, ms = 0.f;
  for (int t = 0; t < TT; ++t) {
    float mf = (float)masks[b * TT + t];
    s  += mf * ctx[((size_t)b * TT + t) * HX + h];
    ms += mf;
  }
  avg[b * HX + h] = s / ms;
}

__global__ void tri_kernel(const float* __restrict__ ctx, const float* __restrict__ avg,
                           const float* __restrict__ tw, const float* __restrict__ tb,
                           float* __restrict__ tri) {
  int m = blockIdx.x * blockDim.x + threadIdx.x;
  if (m >= BT) return;
  int b = m / TT;
  float a0 = tb[0], a1 = tb[1];
  for (int h = 0; h < HX; ++h) {
    float c  = ctx[(size_t)m * HX + h];
    float av = avg[b * HX + h];
    a0 += tw[h] * c + tw[HX + h] * av;
    a1 += tw[2 * HX + h] * c + tw[3 * HX + h] * av;
  }
  tri[m * 2] = a0; tri[m * 2 + 1] = a1;
}

__device__ __forceinline__ float lse2(float x, float y) {
  float m = fmaxf(x, y);
  return m + logf(expf(x - m) + expf(y - m));
}

__global__ void crf_kernel(const float* __restrict__ tri, const int* __restrict__ lens,
                           const float* __restrict__ trans,
                           float* __restrict__ alph, float* __restrict__ sp,
                           float* __restrict__ sps) {
  int b = threadIdx.x;
  if (b >= BB) return;
  int L = lens[b];
  const float t00 = trans[0], t01 = trans[1], t10 = trans[2], t11 = trans[3];
  const float* f = tri + (size_t)b * TT * 2;
  float* A = alph + (size_t)b * TT * 2;
  float a0 = f[0], a1 = f[1];
  A[0] = a0; A[1] = a1;
  for (int t = 1; t < TT; ++t) {
    float n0 = lse2(a0 + t00, a1 + t10) + f[t * 2];
    float n1 = lse2(a0 + t01, a1 + t11) + f[t * 2 + 1];
    if (t < L) { a0 = n0; a1 = n1; }
    A[t * 2] = a0; A[t * 2 + 1] = a1;
  }
  float logZ = lse2(A[(L - 1) * 2], A[(L - 1) * 2 + 1]);
  float ssum = 0.f;
  {
    float v = (TT - 1 < L) ? expf(A[(TT - 1) * 2 + 1] - logZ) : 0.f;
    sp[b * TT + TT - 1] = v; ssum += v;
  }
  float b0 = 0.f, b1 = 0.f;
  for (int t = TT - 2; t >= 0; --t) {
    float fn0 = f[(t + 1) * 2]     + b0;
    float fn1 = f[(t + 1) * 2 + 1] + b1;
    float c0 = lse2(t00 + fn0, t01 + fn1);
    float c1 = lse2(t10 + fn0, t11 + fn1);
    b0 = (t >= L - 1) ? 0.f : c0;
    b1 = (t >= L - 1) ? 0.f : c1;
    float v = (t < L) ? expf(A[t * 2 + 1] + b1 - logZ) : 0.f;
    sp[b * TT + t] = v; ssum += v;
  }
  sps[b] = ssum;
}

__global__ void sentv_kernel(const float* __restrict__ ctx, const float* __restrict__ sp,
                             float* __restrict__ sv) {
  int b = blockIdx.x, h = threadIdx.x;
  float s = 0.f;
  for (int t = 0; t < TT; ++t)
    s += sp[b * TT + t] * ctx[((size_t)b * TT + t) * HX + h];
  sv[b * HX + h] = s;
}

__global__ void head_kernel(const float* __restrict__ sv, const float* __restrict__ sps,
                            const float* __restrict__ lw, const float* __restrict__ lb,
                            const int* __restrict__ labels, float* __restrict__ perb) {
  int b = threadIdx.x;
  if (b >= BB) return;
  float g = sps[b] * 0.5f;
  g = fminf(fmaxf(g, 1e-4f), 1e5f);
  float s[3];
  for (int c = 0; c < 3; ++c) {
    float acc = lb[c];
    for (int h = 0; h < HX; ++h)
      acc += (sv[b * HX + h] / g) * lw[c * HX + h];
    s[c] = acc;
  }
  float m = fmaxf(s[0], fmaxf(s[1], s[2]));
  float Z = m + logf(expf(s[0] - m) + expf(s[1] - m) + expf(s[2] - m));
  perb[b] = s[labels[b]] - Z;
}

__global__ void final_kernel(const float* __restrict__ perb, const float* __restrict__ sps,
                             const float* __restrict__ trans, float* __restrict__ out) {
  float cls = 0.f, sn = 0.f;
  for (int b = 0; b < BB; ++b) { cls += perb[b]; sn += sps[b]; }
  cls = -cls / BB; sn /= BB;
  float p1 = trans[2] - trans[0]; if (p1 < 0.f) p1 = 0.f;
  float p2 = trans[1] - trans[3]; if (p2 < 0.f) p2 = 0.f;
  out[0] = cls;
  out[1] = 0.1f * (p1 + p2) + 0.1f * sn;
}

// =====================================================================
extern "C" void kernel_launch(void* const* d_in, const int* in_sizes, int n_in,
                              void* d_out, int out_size, void* d_ws, size_t ws_size,
                              hipStream_t stream) {
  (void)in_sizes; (void)n_in; (void)out_size; (void)ws_size;
  const int*   sents  = (const int*)d_in[0];
  const int*   lens   = (const int*)d_in[1];
  const int*   masks  = (const int*)d_in[2];
  const int*   labels = (const int*)d_in[3];
  const float* wemb   = (const float*)d_in[4];
  const float* memb   = (const float*)d_in[5];
  const float* wih[4] = {(const float*)d_in[6],  (const float*)d_in[10],
                         (const float*)d_in[14], (const float*)d_in[18]};
  const float* whh[4] = {(const float*)d_in[7],  (const float*)d_in[11],
                         (const float*)d_in[15], (const float*)d_in[19]};
  const float* bih[4] = {(const float*)d_in[8],  (const float*)d_in[12],
                         (const float*)d_in[16], (const float*)d_in[20]};
  const float* bhh[4] = {(const float*)d_in[9],  (const float*)d_in[13],
                         (const float*)d_in[17], (const float*)d_in[21]};
  const float* conv_w = (const float*)d_in[22];
  const float* conv_b = (const float*)d_in[23];
  const float* tri_w  = (const float*)d_in[24];
  const float* tri_b  = (const float*)d_in[25];
  const float* trans  = (const float*)d_in[26];
  const float* lab_w  = (const float*)d_in[27];
  const float* lab_b  = (const float*)d_in[28];
  float* out = (float*)d_out;

  // ---- workspace carve-out (≈110 MB) ----
  char* base = (char*)d_ws; size_t off = 0;
  auto alloc = [&](size_t bytes) -> void* {
    void* p = base + off; off += (bytes + 255) & ~(size_t)255; return p;
  };
  _Float16* X16   = (_Float16*)alloc((size_t)BT * D0P * 2);
  _Float16* X1    = (_Float16*)alloc((size_t)BT * HX * 2);
  _Float16* X2    = (_Float16*)alloc((size_t)BT * HX * 2);
  _Float16* WIH0  = (_Float16*)alloc((size_t)2 * G3 * D0P * 2);
  _Float16* WIH1  = (_Float16*)alloc((size_t)2 * G3 * HX * 2);
  _Float16* WHH16 = (_Float16*)alloc((size_t)4 * G3 * HH * 2);
  _Float16* CW    = (_Float16*)alloc((size_t)3 * HX * HX * 2);
  _Float16* GI    = (_Float16*)alloc((size_t)2 * BT * G3 * 2);
  _Float16* ZROW  = (_Float16*)alloc((size_t)1024 * 2);
  float* GH   = (float*)alloc((size_t)2 * BB * G3 * 4);
  float* HF   = (float*)alloc((size_t)2 * BB * HH * 4);
  float* CTX  = (float*)alloc((size_t)BT * HX * 4);
  float* AVG  = (float*)alloc((size_t)BB * HX * 4);
  float* TRI  = (float*)alloc((size_t)BT * 2 * 4);
  float* ALPH = (float*)alloc((size_t)BT * 2 * 4);
  float* SP   = (float*)alloc((size_t)BT * 4);
  float* SPS  = (float*)alloc((size_t)BB * 4);
  float* PERB = (float*)alloc((size_t)BB * 4);
  float* SV   = (float*)alloc((size_t)BB * HX * 4);
  int* RM  = (int*)alloc((size_t)BT * 4);
  int* CM0 = (int*)alloc((size_t)BT * 4);
  int* CM2 = (int*)alloc((size_t)BT * 4);

  auto g1 = [](long n) { return (unsigned)((n + 255) / 256); };

  // ---- weight packing (f32 -> f16, pad K to 32) + zero row ----
  zero16_kernel<<<g1(1024), 256, 0, stream>>>(ZROW, 1024);
  pack16_kernel<<<g1((long)G3 * D0P), 256, 0, stream>>>(wih[0], WIH0,              G3, 350, D0P);
  pack16_kernel<<<g1((long)G3 * D0P), 256, 0, stream>>>(wih[1], WIH0 + G3 * D0P,   G3, 350, D0P);
  pack16_kernel<<<g1((long)G3 * HX),  256, 0, stream>>>(wih[2], WIH1,              G3, HX,  HX);
  pack16_kernel<<<g1((long)G3 * HX),  256, 0, stream>>>(wih[3], WIH1 + G3 * HX,    G3, HX,  HX);
  for (int d = 0; d < 4; ++d)
    pack16_kernel<<<g1((long)G3 * HH), 256, 0, stream>>>(whh[d], WHH16 + (size_t)d * G3 * HH, G3, HH, HH);
  packconv_kernel<<<g1(3L * HX * HX), 256, 0, stream>>>(conv_w, CW);

  embed_kernel<<<g1((long)BT * D0P), 256, 0, stream>>>(sents, masks, wemb, memb, X16);
  build_maps_kernel<<<g1(BT), 256, 0, stream>>>(lens, RM, CM0, CM2);

  const int NT2_G = G3 / 32;           // 24
  const int numG  = (BT / 32) * NT2_G; // 9600
  // ---- layer 0: input projections (fwd identity, bwd via reverse rowmap) ----
  wmma_gemm_kernel<<<numG / 8, 256, 0, stream>>>(X16, D0P, WIH0,            D0P,
      (float*)0, GI,                   G3, (int*)0, ZROW, NT2_G, numG, D0P / 32, 0);
  wmma_gemm_kernel<<<numG / 8, 256, 0, stream>>>(X16, D0P, WIH0 + G3 * D0P, D0P,
      (float*)0, GI + (size_t)BT * G3, G3, RM,      ZROW, NT2_G, numG, D0P / 32, 0);
  gru_layer_kernel<<<2, 1024, BB * HH * 2, stream>>>(GI, WHH16,
      bih[0], bhh[0], bih[1], bhh[1], lens, X1, GH, HF);

  // ---- layer 1 ----
  wmma_gemm_kernel<<<numG / 8, 256, 0, stream>>>(X1, HX, WIH1,           HX,
      (float*)0, GI,                   G3, (int*)0, ZROW, NT2_G, numG, HX / 32, 0);
  wmma_gemm_kernel<<<numG / 8, 256, 0, stream>>>(X1, HX, WIH1 + G3 * HX, HX,
      (float*)0, GI + (size_t)BT * G3, G3, RM,      ZROW, NT2_G, numG, HX / 32, 0);
  gru_layer_kernel<<<2, 1024, BB * HH * 2, stream>>>(GI, WHH16 + (size_t)2 * G3 * HH,
      bih[2], bhh[2], bih[3], bhh[3], lens, X2, GH, HF);

  // ---- conv as 3 shifted GEMMs (rowmap handles zero padding) ----
  const int NT2_C = HX / 32;           // 16
  const int numC  = (BT / 32) * NT2_C; // 6400
  wmma_gemm_kernel<<<numC / 8, 256, 0, stream>>>(X2, HX, CW,               HX,
      CTX, (_Float16*)0, HX, CM0,     ZROW, NT2_C, numC, HX / 32, 0);
  wmma_gemm_kernel<<<numC / 8, 256, 0, stream>>>(X2, HX, CW + HX * HX,     HX,
      CTX, (_Float16*)0, HX, (int*)0, ZROW, NT2_C, numC, HX / 32, 1);
  wmma_gemm_kernel<<<numC / 8, 256, 0, stream>>>(X2, HX, CW + 2 * HX * HX, HX,
      CTX, (_Float16*)0, HX, CM2,     ZROW, NT2_C, numC, HX / 32, 1);
  conv_epilogue_kernel<<<g1((long)BT * HX), 256, 0, stream>>>(CTX, conv_b);

  // ---- pooling, CRF, heads ----
  avg_kernel<<<BB, HX, 0, stream>>>(CTX, masks, AVG);
  tri_kernel<<<g1(BT), 256, 0, stream>>>(CTX, AVG, tri_w, tri_b, TRI);
  crf_kernel<<<1, 128, 0, stream>>>(TRI, lens, trans, ALPH, SP, SPS);
  sentv_kernel<<<BB, HX, 0, stream>>>(CTX, SP, SV);
  head_kernel<<<1, 128, 0, stream>>>(SV, SPS, lab_w, lab_b, labels, PERB);
  final_kernel<<<1, 1, 0, stream>>>(PERB, SPS, trans, out);
}
